// Method_24781961298221
// MI455X (gfx1250) — matrix operand order repaired and checked by reading, not compiled, not run on hardware
//
#include <hip/hip_runtime.h>
#include <stdint.h>

typedef _Float16 v16h __attribute__((ext_vector_type(16)));
typedef _Float16 v8h  __attribute__((ext_vector_type(8)));
typedef float    v8f  __attribute__((ext_vector_type(8)));
typedef float    vf4  __attribute__((ext_vector_type(4)));
typedef unsigned int u32x4 __attribute__((ext_vector_type(4)));
typedef int      i32x8 __attribute__((ext_vector_type(8)));
typedef int      i32x4 __attribute__((ext_vector_type(4)));

#define T_DIM  2048
#define B_DIM  16
#define ID_DIM 1024
#define HD_DIM 1024
#define M_DIM  (T_DIM * B_DIM)   // 32768
#define N_DIM  (3 * HD_DIM)      // 3072
#define K_DIM  ID_DIM            // 1024
#define KTILE  32

#define SHUF16(a, b)                                                          \
    __builtin_shufflevector(a, b, 0, 1, 2, 3, 4, 5, 6, 7, 8, 9, 10, 11, 12,  \
                            13, 14, 15)

#if __has_builtin(__builtin_amdgcn_tensor_load_to_lds)
#define HAS_TDM 1
#else
#define HAS_TDM 0
#endif

#if HAS_TDM
// ---------------------------------------------------------------------------
// Issue one TDM 2D tile load: global (row-major, K contiguous) -> LDS.
// Tile = 128 rows x KTILE elements, f16 (data_size code 1 = 2 bytes).
// D# layout per CDNA5 ISA 8.3-8.5. Groups 2/3 zero (2D tensor).
// ---------------------------------------------------------------------------
__device__ __forceinline__ void tdm_load_tile(const _Float16* gtile,
                                              uint32_t lds_off,
                                              uint32_t tensor_rows)
{
    const uint64_t ga = (uint64_t)(uintptr_t)gtile;
    u32x4 g0;
    g0[0] = 1u;                                  // count=1 (valid), user D#
    g0[1] = lds_off;                             // lds_addr (bytes)
    g0[2] = (uint32_t)ga;                        // global_addr[31:0]
    g0[3] = (uint32_t)((ga >> 32) & 0x01FFFFFFu) // global_addr[56:32]
          | (2u << 30);                          // type = 2 ("image")

    i32x8 g1;
    g1[0] = (int)(1u << 16);                     // data_size=1 (2B); no mask
    g1[1] = (int)(((uint32_t)K_DIM & 0xFFFFu) << 16);  // tensor_dim0[15:0]
    g1[2] = (int)(((uint32_t)K_DIM >> 16) & 0xFFFFu)   // tensor_dim0[31:16]
          | (int)((tensor_rows & 0xFFFFu) << 16);      // tensor_dim1[15:0]
    g1[3] = (int)((tensor_rows >> 16) & 0xFFFFu)       // tensor_dim1[31:16]
          | (int)((uint32_t)KTILE << 16);              // tile_dim0 = 32
    g1[4] = 128;                                 // tile_dim1=128, tile_dim2=0
    g1[5] = K_DIM;                               // tensor_dim0_stride[31:0]
    g1[6] = 0;                                   // stride hi, dim1_stride lo
    g1[7] = 0;

    i32x4 gz = {0, 0, 0, 0};
#if __clang_major__ >= 23
    i32x8 gz8 = {0, 0, 0, 0, 0, 0, 0, 0};
    __builtin_amdgcn_tensor_load_to_lds(g0, g1, gz, gz, gz8, 0);
#else
    __builtin_amdgcn_tensor_load_to_lds(g0, g1, gz, gz, 0);
#endif
}
__device__ __forceinline__ void wait_tensor_le2()
{
#if __has_builtin(__builtin_amdgcn_s_wait_tensorcnt)
    __builtin_amdgcn_s_wait_tensorcnt((short)2);
#else
    asm volatile("s_wait_tensorcnt 0x2" ::: "memory");
#endif
}
__device__ __forceinline__ void wait_tensor_0()
{
#if __has_builtin(__builtin_amdgcn_s_wait_tensorcnt)
    __builtin_amdgcn_s_wait_tensorcnt((short)0);
#else
    asm volatile("s_wait_tensorcnt 0x0" ::: "memory");
#endif
}
#else
// Fallback staging: regular global load + ds_store, 16 f16 per thread.
__device__ __forceinline__ void cp16(const _Float16* g, _Float16* l)
{
    const v8h* gs = (const v8h*)g;
    v8h*       ls = (v8h*)l;
    ls[0] = gs[0];
    ls[1] = gs[1];
}
#endif

// ---------------------------------------------------------------------------
// Pre-conversion: fp32 -> fp16 once, so the GEMM loop has zero v_cvt.
// ---------------------------------------------------------------------------
__global__ __launch_bounds__(256)
void f32_to_f16(const float* __restrict__ src, _Float16* __restrict__ dst,
                size_t n8)
{
    size_t i = (size_t)blockIdx.x * blockDim.x + threadIdx.x;
    if (i >= n8) return;
    const vf4* p = (const vf4*)(src + i * 8);
    vf4 a = p[0], b = p[1];
    v8h o;
#pragma unroll
    for (int j = 0; j < 4; ++j) {
        o[j]     = (_Float16)a[j];
        o[4 + j] = (_Float16)b[j];
    }
    *(v8h*)(dst + i * 8) = o;
}

// ---------------------------------------------------------------------------
// GEMM: proj[m, n] = sum_k x[m, k] * W[n, k] + bias[n]   (f16 in, f32 out)
// Block tile 128x128 (8 waves as 4(M) x 2(N), each wave 32x64 -> 8 WMMAs/step)
// K staged in 32-wide double-buffered LDS tiles by the Tensor Data Mover:
//   wave0: tensor_load_to_lds(next A), tensor_load_to_lds(next B),
//          s_wait_tensorcnt 2 (retire current pair only) -> barrier ->
//   all waves: ds_load fragments + 8 x v_wmma -> barrier.
// Fragment layouts per CDNA5 ISA (16-bit A 16x32, B 32x16, f32 C/D 16x16).
// ---------------------------------------------------------------------------
__global__ __launch_bounds__(256)
void proj_gemm_wmma(const _Float16* __restrict__ Xh,
                    const _Float16* __restrict__ Wh,
                    const float* __restrict__ bias, float* __restrict__ P)
{
    __shared__ _Float16 Abuf[2][128 * KTILE];   // [row][k], 64B rows
    __shared__ _Float16 Bbuf[2][128 * KTILE];

    const int tid  = threadIdx.x;
    const int lane = tid & 31;
    const int wave = tid >> 5;
    const int half = lane >> 4;   // half-wave selects K sub-range
    const int lrow = lane & 15;   // M index for A, N index for B/C/D
    const int wm   = wave & 3;
    const int wn   = wave >> 2;

    const int m0 = blockIdx.y * 128;
    const int n0 = blockIdx.x * 128;

    v8f acc[2][4];
#pragma unroll
    for (int mt = 0; mt < 2; ++mt)
#pragma unroll
        for (int nt = 0; nt < 4; ++nt) acc[mt][nt] = v8f{};

#if HAS_TDM
    const uint32_t ldsA[2] = {(uint32_t)(uintptr_t)(&Abuf[0][0]),
                              (uint32_t)(uintptr_t)(&Abuf[1][0])};
    const uint32_t ldsB[2] = {(uint32_t)(uintptr_t)(&Bbuf[0][0]),
                              (uint32_t)(uintptr_t)(&Bbuf[1][0])};
    const _Float16* gAblk = Xh + (size_t)m0 * K_DIM;
    const _Float16* gBblk = Wh + (size_t)n0 * K_DIM;
    if (wave == 0) {                 // one TDM issue pair per tile
        tdm_load_tile(gAblk, ldsA[0], M_DIM);
        tdm_load_tile(gBblk, ldsB[0], N_DIM);
    }
#else
    const int srow   = tid >> 1;        // 0..127
    const int schunk = (tid & 1) * 16;  // element offset 0 / 16
    const _Float16* gA = Xh + (size_t)(m0 + srow) * K_DIM + schunk;
    const _Float16* gB = Wh + (size_t)(n0 + srow) * K_DIM + schunk;
    const int soff = srow * KTILE + schunk;
    cp16(gA, &Abuf[0][soff]);
    cp16(gB, &Bbuf[0][soff]);
#endif

    int buf = 0;
    for (int k = 0; k < K_DIM; k += KTILE) {
#if HAS_TDM
        if (wave == 0) {
            if (k + KTILE < K_DIM) {
                // Stage next tile while current is consumed.
                tdm_load_tile(gAblk + (k + KTILE), ldsA[buf ^ 1], M_DIM);
                tdm_load_tile(gBblk + (k + KTILE), ldsB[buf ^ 1], N_DIM);
                wait_tensor_le2();   // oldest pair (current tile) landed
            } else {
                wait_tensor_0();
            }
        }
#else
        if (k + KTILE < K_DIM) {
            cp16(gA + (k + KTILE), &Abuf[buf ^ 1][soff]);
            cp16(gB + (k + KTILE), &Bbuf[buf ^ 1][soff]);
        }
#endif
        __syncthreads();        // staged data visible to all waves

        // A fragments (16-bit A 16x32): lanes 0-15 M=lane; j<8 -> K=j(+8h),
        // j>=8 -> K=8+j(+8h); two 16B LDS chunks per fragment.
        const _Float16* la0 = &Abuf[buf][(wm * 32 + lrow) * KTILE];
        const _Float16* la1 = la0 + 16 * KTILE;
        v16h af[2];
        {
            const v8h* p0 = (const v8h*)(la0 + 8 * half);
            const v8h* p1 = (const v8h*)(la0 + 16 + 8 * half);
            af[0] = SHUF16(p0[0], p1[0]);
            const v8h* q0 = (const v8h*)(la1 + 8 * half);
            const v8h* q1 = (const v8h*)(la1 + 16 + 8 * half);
            af[1] = SHUF16(q0[0], q1[0]);
        }
        // B fragments (32x16): N = lane%16, K same order as A (8*half, 16+8*half).
        v16h bf[4];
#pragma unroll
        for (int nt = 0; nt < 4; ++nt) {
            const _Float16* lb =
                &Bbuf[buf][(wn * 64 + nt * 16 + lrow) * KTILE];
            const v8h* p0 = (const v8h*)(lb + 8 * half);
            const v8h* p1 = (const v8h*)(lb + 16 + 8 * half);
            bf[nt] = SHUF16(p0[0], p1[0]);
        }

#pragma unroll
        for (int mt = 0; mt < 2; ++mt)
#pragma unroll
            for (int nt = 0; nt < 4; ++nt)
                acc[mt][nt] = __builtin_amdgcn_wmma_f32_16x16x32_f16(
                    false, af[mt], false, bf[nt], (short)0, acc[mt][nt], false,
                    false);

        __syncthreads();        // done reading buf before it is restaged
        buf ^= 1;
    }

    // Store: f32 C/D VGPR r -> M = r + 8*half, N = lane%16; fuse bias add.
#pragma unroll
    for (int nt = 0; nt < 4; ++nt) {
        const int nc = n0 + wn * 64 + nt * 16 + lrow;
        const float bv = bias[nc];
#pragma unroll
        for (int mt = 0; mt < 2; ++mt) {
#pragma unroll
            for (int r = 0; r < 8; ++r) {
                P[(size_t)(m0 + wm * 32 + mt * 16 + r + 8 * half) * N_DIM +
                  nc] = acc[mt][nt][r] + bv;
            }
        }
    }
}

// ---------------------------------------------------------------------------
// Fused elementwise scan: each (b,h) channel is an independent length-T
// recurrence -> one thread per channel, S in a register, coalesced streaming.
// ---------------------------------------------------------------------------
__global__ __launch_bounds__(256)
void lstm_scan(const float* __restrict__ proj, const float* __restrict__ x,
               const float* __restrict__ S0,
               const float* __restrict__ uf, const float* __restrict__ bfv,
               const float* __restrict__ ur, const float* __restrict__ brv,
               const float* __restrict__ PFB, const float* __restrict__ HWb,
               float* __restrict__ Y, float* __restrict__ Qf,
               float* __restrict__ Sf)
{
    const int i = blockIdx.x * blockDim.x + threadIdx.x; // 0 .. B*HD-1
    const int h = i & (HD_DIM - 1);
    const int b = i >> 10; // HD = 1024

    float S = S0[i];
    const float ufv  = uf[h];
    const float fsum = bfv[h] + PFB[h];
    const float urv  = ur[h];
    const float rsum = brv[h] + HWb[h];

    const size_t strideP = (size_t)B_DIM * N_DIM;
    const size_t strideX = (size_t)B_DIM * HD_DIM;

    const float* pF = proj + (size_t)b * N_DIM + h;
    const float* pR = pF + HD_DIM;
    const float* pA = pR + HD_DIM;
    const float* pX = x + i;
    float*       pY = Y + i;

    float Q = 0.0f;
#pragma unroll 2
    for (int t = 0; t < T_DIM; ++t) {
        const float fi = *pF, ri = *pR, ai = *pA, xk = *pX;
        const float F = 1.0f / (1.0f + __expf(-(fi + ufv * S + fsum)));
        const float R = 1.0f / (1.0f + __expf(-(ri + urv * S + rsum)));
        S = F * S + (1.0f - F) * ai;
        Q = R * S + (1.0f - R) * xk;
        *pY = Q;
        pF += strideP; pR += strideP; pA += strideP;
        pX += strideX; pY += strideX;
    }
    Qf[i] = Q;
    Sf[i] = S;
}

// ---------------------------------------------------------------------------
extern "C" void kernel_launch(void* const* d_in, const int* in_sizes, int n_in,
                              void* d_out, int out_size, void* d_ws,
                              size_t ws_size, hipStream_t stream)
{
    const float* x    = (const float*)d_in[0];
    // d_in[1] = Q0 (unused by the reference computation)
    const float* S0   = (const float*)d_in[2];
    const float* W_ih = (const float*)d_in[3];
    const float* b_ih = (const float*)d_in[4];
    const float* uf   = (const float*)d_in[5];
    const float* bf   = (const float*)d_in[6];
    const float* ur   = (const float*)d_in[7];
    const float* br   = (const float*)d_in[8];
    const float* PFB  = (const float*)d_in[9];
    const float* HWb  = (const float*)d_in[10];

    // Workspace: xh (M*K f16, 64MB) | wh (N*K f16, 6MB) | proj (M*N f32, 384MB)
    _Float16* xh   = (_Float16*)d_ws;
    _Float16* wh   = xh + (size_t)M_DIM * K_DIM;
    float*    proj = (float*)(wh + (size_t)N_DIM * K_DIM);

    float* Y  = (float*)d_out;                       // [T, B, HD]
    float* Qf = Y + (size_t)T_DIM * B_DIM * HD_DIM;  // [1, B, HD]
    float* Sf = Qf + (size_t)B_DIM * HD_DIM;         // [1, B, HD]

    const size_t x_n8 = (size_t)M_DIM * K_DIM / 8;
    const size_t w_n8 = (size_t)N_DIM * K_DIM / 8;
    f32_to_f16<<<(unsigned)((x_n8 + 255) / 256), dim3(256), 0, stream>>>(
        x, xh, x_n8);
    f32_to_f16<<<(unsigned)((w_n8 + 255) / 256), dim3(256), 0, stream>>>(
        W_ih, wh, w_n8);

    dim3 gemm_grid(N_DIM / 128, M_DIM / 128);
    proj_gemm_wmma<<<gemm_grid, dim3(256), 0, stream>>>(xh, wh, b_ih, proj);

    lstm_scan<<<(B_DIM * HD_DIM) / 256, dim3(256), 0, stream>>>(
        proj, x, S0, uf, bf, ur, br, PFB, HWb, Y, Qf, Sf);
}
